// AttentionLayer_86586540687923
// MI455X (gfx1250) — matrix operand order repaired and checked by reading, not compiled, not run on hardware
//
#include <hip/hip_runtime.h>
#include <hip/hip_bf16.h>

#define B_ 32
#define T_ 2048
#define D_ 1024
#define M_ 128

typedef __attribute__((ext_vector_type(16))) __bf16 v16bf;
typedef __attribute__((ext_vector_type(8)))  float  v8f;

union BV { uint4 u[2]; v16bf v; };

// ---------------------------------------------------------------------------
// Kernel 0: pack W [D,M] f32 (row-major, stride M) into bf16 in WMMA B-matrix
// order, stored as two 512B "planes" per 32x16 tile so LDS reads are
// conflict-free:
//   tile t = kc*8 + mt; plane p = e>>3 (elements 0-7 vs 8-15)
//   Wp[((t*2 + p)*32 + lane)*8 + (e&7)]
// Lane L holds column n = L&15; element e holds K = 16*(L>>4) + e.
// Each (plane, lane) is one contiguous uint4 (8 bf16), lane stride 16B.
// ---------------------------------------------------------------------------
__global__ void __launch_bounds__(256)
pack_w_kernel(const float* __restrict__ W, __bf16* __restrict__ Wp) {
    int idx  = blockIdx.x * blockDim.x + threadIdx.x;   // 0 .. D*M-1 (out idx)
    int e8   = idx & 7;
    int lane = (idx >> 3) & 31;
    int p    = (idx >> 8) & 1;
    int tile = idx >> 9;           // 0..255
    int mt   = tile & 7;
    int kc   = tile >> 3;          // 0..31
    int e    = p * 8 + e8;
    int k = kc * 32 + ((lane >> 4) << 4) + e;
    int m = mt * 16 + (lane & 15);
    Wp[idx] = (__bf16)W[k * M_ + m];
}

// ---------------------------------------------------------------------------
// Kernel 1: per-token score utu[b,t] = us . tanh(W^T x[b,t] + bias)
// 512-thread block = 16 wave32s; the whole packed W (256 KB) is staged into
// LDS once per block (CDNA5: 320 KB LDS per workgroup). Per K-step, all 8
// B tiles are fetched into registers as one ds_load clause BEFORE the 8
// WMMAs so LDS latency overlaps the matrix pipe (avoids s_wait_dscnt 0x0
// in front of every wmma).
// A layout (16x32 bf16): row = lane&15; lanes 0-15: e0..15 -> K 0..15;
// lanes 16-31: e0..15 -> K 16..31.
// ---------------------------------------------------------------------------
__global__ void __launch_bounds__(512)
score_kernel(const float* __restrict__ x, const __bf16* __restrict__ Wp,
             const float* __restrict__ bias, const float* __restrict__ us,
             float* __restrict__ utu) {
    extern __shared__ uint4 lds[];                 // 256 KB = 16384 uint4

    // ---- cooperative one-time stage of packed W into LDS ----
    {
        const uint4* g = (const uint4*)Wp;
        #pragma unroll 4
        for (int i = threadIdx.x; i < (D_ * M_ * 2) / 16; i += 512) {
            lds[i] = g[i];
        }
    }
    __syncthreads();

    const int lane = threadIdx.x & 31;
    const int wave = (blockIdx.x * blockDim.x + threadIdx.x) >> 5; // token group
    const int row  = lane & 15;
    const int half = lane >> 4;
    const long tok0 = (long)wave * 16;
    const float* __restrict__ xrow = x + (tok0 + row) * D_;

    v8f acc[8] = {};

    for (int kc = 0; kc < D_ / 32; ++kc) {
        const int k0 = kc * 32 + half * 16;
        float4 q0 = *(const float4*)(xrow + k0);
        float4 q1 = *(const float4*)(xrow + k0 + 4);
        float4 q2 = *(const float4*)(xrow + k0 + 8);
        float4 q3 = *(const float4*)(xrow + k0 + 12);
        if (kc + 1 < D_ / 32) {
            __builtin_prefetch(xrow + k0 + 32, 0, 3);   // global_prefetch_b8
        }

        // fetch ALL 8 B tiles first (one ds_load_b128 clause, 16 loads)
        BV bv[8];
        #pragma unroll
        for (int mt = 0; mt < 8; ++mt) {
            const int tb = (kc * 8 + mt) * 64 + lane;   // uint4 units
            bv[mt].u[0] = lds[tb];        // plane 0: elements 0-7
            bv[mt].u[1] = lds[tb + 32];   // plane 1: elements 8-15
        }

        v16bf av;
        av[0]  = (__bf16)q0.x; av[1]  = (__bf16)q0.y;
        av[2]  = (__bf16)q0.z; av[3]  = (__bf16)q0.w;
        av[4]  = (__bf16)q1.x; av[5]  = (__bf16)q1.y;
        av[6]  = (__bf16)q1.z; av[7]  = (__bf16)q1.w;
        av[8]  = (__bf16)q2.x; av[9]  = (__bf16)q2.y;
        av[10] = (__bf16)q2.z; av[11] = (__bf16)q2.w;
        av[12] = (__bf16)q3.x; av[13] = (__bf16)q3.y;
        av[14] = (__bf16)q3.z; av[15] = (__bf16)q3.w;

        #pragma unroll
        for (int mt = 0; mt < 8; ++mt) {
            acc[mt] = __builtin_amdgcn_wmma_f32_16x16x32_bf16(
                false, av, false, bv[mt].v, (short)0, acc[mt], false, false);
        }
    }

    // Epilogue: lane column m = mt*16 + (lane&15); C/D layout:
    // element j of v8f in lane L is token (j + 8*(L>>4)), column L&15.
    float bb[8], uu[8];
    #pragma unroll
    for (int mt = 0; mt < 8; ++mt) {
        int m = mt * 16 + row;
        bb[mt] = bias[m];
        uu[mt] = us[m];
    }
    float pj[8];
    #pragma unroll
    for (int j = 0; j < 8; ++j) {
        float p = 0.f;
        #pragma unroll
        for (int mt = 0; mt < 8; ++mt) {
            p += tanhf(acc[mt][j] + bb[mt]) * uu[mt];
        }
        pj[j] = p;
    }
    // reduce the 16 columns inside each half-wave (xor masks 1,2,4,8 keep
    // lanes 0-15 and 16-31 separate -> token halves stay separate)
    #pragma unroll
    for (int j = 0; j < 8; ++j) {
        #pragma unroll
        for (int msk = 1; msk <= 8; msk <<= 1) {
            pj[j] += __shfl_xor(pj[j], msk, 32);
        }
    }
    if (row == 0) {
        #pragma unroll
        for (int j = 0; j < 8; ++j) {
            utu[tok0 + half * 8 + j] = pj[j];
        }
    }
}

// ---------------------------------------------------------------------------
// Kernel 2: masked softmax over T per batch row (exact reference semantics:
// max over all t, mask applied to e after exp).
// ---------------------------------------------------------------------------
__global__ void __launch_bounds__(256)
softmax_kernel(const float* __restrict__ utu,
               const unsigned char* __restrict__ mask,
               float* __restrict__ a) {
    __shared__ float red[256];
    const int b   = blockIdx.x;
    const int tid = threadIdx.x;

    float v[8];
    float mx = -INFINITY;
    #pragma unroll
    for (int i = 0; i < 8; ++i) {
        v[i] = utu[b * T_ + tid + i * 256];
        mx = fmaxf(mx, v[i]);
    }
    red[tid] = mx;
    __syncthreads();
    for (int s = 128; s > 0; s >>= 1) {
        if (tid < s) red[tid] = fmaxf(red[tid], red[tid + s]);
        __syncthreads();
    }
    mx = red[0];
    __syncthreads();

    float e[8];
    float sum = 0.f;
    #pragma unroll
    for (int i = 0; i < 8; ++i) {
        float msk = mask[b * T_ + tid + i * 256] ? 1.f : 0.f;
        e[i] = __expf(v[i] - mx) * msk;
        sum += e[i];
    }
    red[tid] = sum;
    __syncthreads();
    for (int s = 128; s > 0; s >>= 1) {
        if (tid < s) red[tid] += red[tid + s];
        __syncthreads();
    }
    const float inv = 1.0f / red[0];
    #pragma unroll
    for (int i = 0; i < 8; ++i) {
        a[b * T_ + tid + i * 256] = e[i] * inv;
    }
}

// ---------------------------------------------------------------------------
// Kernel 3a: partial weighted sums. Block = (b, t-chunk of 128); each thread
// owns one float4 of D. Deterministic two-stage reduction (no atomics).
// ---------------------------------------------------------------------------
__global__ void __launch_bounds__(256)
wsum_partial_kernel(const float* __restrict__ x, const float* __restrict__ a,
                    float* __restrict__ part) {
    const int b   = blockIdx.x >> 4;
    const int tc  = blockIdx.x & 15;
    const int tid = threadIdx.x;       // float4 index within D
    float4 acc = {0.f, 0.f, 0.f, 0.f};
    const int t0 = tc * 128;
    for (int t = t0; t < t0 + 128; ++t) {
        const float w = a[b * T_ + t];
        const float4 xv =
            ((const float4*)(x + (long)(b * T_ + t) * D_))[tid];
        acc.x += w * xv.x;
        acc.y += w * xv.y;
        acc.z += w * xv.z;
        acc.w += w * xv.w;
    }
    ((float4*)(part + (long)(b * 16 + tc) * D_))[tid] = acc;
}

// ---------------------------------------------------------------------------
// Kernel 3b: reduce the 16 partials -> out[b,d]
// ---------------------------------------------------------------------------
__global__ void __launch_bounds__(256)
wsum_final_kernel(const float* __restrict__ part, float* __restrict__ out) {
    const int b   = blockIdx.x;
    const int tid = threadIdx.x;
    float4 acc = {0.f, 0.f, 0.f, 0.f};
    #pragma unroll
    for (int tc = 0; tc < 16; ++tc) {
        const float4 p = ((const float4*)(part + (long)(b * 16 + tc) * D_))[tid];
        acc.x += p.x;
        acc.y += p.y;
        acc.z += p.z;
        acc.w += p.w;
    }
    ((float4*)(out + (long)b * D_))[tid] = acc;
}

// ---------------------------------------------------------------------------
extern "C" void kernel_launch(void* const* d_in, const int* in_sizes, int n_in,
                              void* d_out, int out_size, void* d_ws,
                              size_t ws_size, hipStream_t stream) {
    const float*         x    = (const float*)d_in[0];
    const float*         W    = (const float*)d_in[1];
    const float*         bias = (const float*)d_in[2];
    const float*         us   = (const float*)d_in[3];
    const unsigned char* mask = (const unsigned char*)d_in[4];
    float* out = (float*)d_out;

    char* ws = (char*)d_ws;
    float*  utu  = (float*)(ws + 0);             // B*T f32   = 256 KB
    float*  a    = (float*)(ws + (256u << 10));  // B*T f32   = 256 KB
    __bf16* Wp   = (__bf16*)(ws + (512u << 10)); // D*M bf16  = 256 KB
    float*  part = (float*)(ws + (768u << 10));  // B*16*D f32 = 2 MB

    pack_w_kernel<<<(D_ * M_) / 256, 256, 0, stream>>>(W, Wp);

    // B*T/16 = 4096 token-group waves, 16 waves (512 thr) per block -> 256
    // blocks; 256 KB dynamic LDS per workgroup holds the full packed W
    // (CDNA5 allows up to 320 KB per workgroup).
    score_kernel<<<(B_ * T_ / 16) / 16, 512, 256 * 1024, stream>>>(
        x, Wp, bias, us, utu);

    softmax_kernel<<<B_, 256, 0, stream>>>(utu, mask, a);

    wsum_partial_kernel<<<B_ * 16, 256, 0, stream>>>(x, a, part);
    wsum_final_kernel<<<B_, 256, 0, stream>>>(part, out);
}
